// spk_vq_vae_2L_lstm_resnet_50294067036782
// MI455X (gfx1250) — compile-verified, matrix operand order checked by reading, not run on hardware
//
#include <hip/hip_runtime.h>
#include <hip/hip_fp16.h>

// ---------------------------------------------------------------------------
// VQ-VAE forward for MI455X (gfx1250, wave32). All matmul-shaped work runs
// through v_wmma_f32_16x16x32_f16 (f32 accumulate). Activations feeding WMMA
// are kept as f16 channels-last companions so every A fragment is two
// global_load_b128 (two 8-wide ci runs per lane); weights are pre-packed into
// the B-fragment layout. OOB conv taps resolve to a zero guard row via
// pointer select (branchless, EXEC stays all-1s around WMMA).
// ---------------------------------------------------------------------------

typedef __attribute__((ext_vector_type(16))) _Float16 v16h;
typedef __attribute__((ext_vector_type(8)))  _Float16 v8h;
typedef __attribute__((ext_vector_type(8)))  float    v8f;

#define DEVFN __device__ __forceinline__

__host__ __device__ constexpr int clog2(int v) {
  int r = 0;
  while ((1 << r) < v) ++r;
  return r;
}

// K index inside a 16x32 f16 A/B fragment, per CDNA5 ISA 7.12.2:
// lanes 0-15 cover K {0..7,16..23}, lanes 16-31 cover K {8..15,24..31}.
DEVFN int frag_k(int lane, int h) {
  int v = h >> 1;
  int i = h & 1;
  return ((v >> 2) << 4) + ((lane >> 4) << 3) + ((v & 3) << 1) + i;
}

// ---------------------------------------------------------------------------
// Weight packing into B-fragment layout, fragment index
// ((ntile*nKc+chunk)*32+lane)*16+h.  K ordering: kk = k*Cin + ci.
// mode 0: normal conv, W[co,ci,k];  mode 1: transposed conv, W[ci,co,k]
// ---------------------------------------------------------------------------
__global__ __launch_bounds__(256) void pack_conv_weights(
    const float* __restrict__ W, _Float16* __restrict__ packB,
    int Cout, int Cin, int KW, int mode) {
  int Ktot = Cin * KW;
  int nKc = (Ktot + 31) >> 5;
  int nNt = (Cout + 15) >> 4;
  int idx = blockIdx.x * blockDim.x + threadIdx.x;
  int total = nNt * nKc * 512;
  if (idx >= total) return;
  int h = idx & 15;
  int lane = (idx >> 4) & 31;
  int rest = idx >> 9;
  int chunk = rest % nKc;
  int ntile = rest / nKc;
  int kk = chunk * 32 + frag_k(lane, h);
  int n = ntile * 16 + (lane & 15);
  float val = 0.f;
  if (kk < Ktot && n < Cout) {
    int ci = kk % Cin;   // kk = k*Cin + ci
    int k = kk / Cin;
    val = (mode == 0) ? W[(n * Cin + ci) * KW + k]
                      : W[(ci * Cout + n) * KW + k];
  }
  packB[idx] = (_Float16)val;
}

__global__ __launch_bounds__(256) void pack_lstm_weights(
    const float* __restrict__ wih, const float* __restrict__ whh,
    _Float16* __restrict__ packB, int Cin, int Hh) {
  int N = 4 * Hh;
  int Ktot = Cin + Hh;
  int nKc = (Ktot + 31) >> 5;
  int nNt = N >> 4;
  int idx = blockIdx.x * blockDim.x + threadIdx.x;
  int total = nNt * nKc * 512;
  if (idx >= total) return;
  int h = idx & 15;
  int lane = (idx >> 4) & 31;
  int rest = idx >> 9;
  int chunk = rest % nKc;
  int ntile = rest / nKc;
  int kk = chunk * 32 + frag_k(lane, h);
  int n = ntile * 16 + (lane & 15);
  float val = 0.f;
  if (kk < Ktot)
    val = (kk < Cin) ? wih[n * Cin + kk] : whh[n * Hh + (kk - Cin)];
  packB[idx] = (_Float16)val;
}

// [B,C,T] f32 -> [B,T,C] f16 (channels-last companion of the network input)
__global__ __launch_bounds__(256) void convert_nct_ntc(
    const float* __restrict__ X, _Float16* __restrict__ Xh, int cLog,
    int tLog, long total) {
  long i = blockIdx.x * (long)blockDim.x + threadIdx.x;
  if (i >= total) return;
  int t = (int)i & ((1 << tLog) - 1);
  int c = (int)(i >> tLog) & ((1 << cLog) - 1);
  long b = i >> (tLog + cLog);
  Xh[((size_t)b << (tLog + cLog)) + ((size_t)t << cLog) + c] =
      (_Float16)X[i];
}

// ---------------------------------------------------------------------------
// Conv / convT as implicit GEMM, fully specialized per layer shape.
// One wave per 16(M=b*T) x 16(N=Cout) tile. A from f16 [B,TIN,CIN]:
// two 8-wide ci runs per lane per chunk = two b128 loads.
// ---------------------------------------------------------------------------
template <int CIN, int KW, int STRIDE, int PAD, int TRANS, int TIN, int TOUT,
          int COUT, int WH>
__global__ __launch_bounds__(32) void conv_gemm_wmma(
    const _Float16* __restrict__ Xh, const _Float16* __restrict__ zrow,
    const _Float16* __restrict__ packB, const float* __restrict__ bias,
    float* __restrict__ Y, _Float16* __restrict__ Yh) {
  constexpr int CLOG = clog2(CIN);
  constexpr int TLOG = clog2(TOUT);
  constexpr int Ktot = CIN * KW;           // multiple of 32 for all layers
  constexpr int nKc = Ktot / 32;

  int lane = threadIdx.x & 31;
  int row = lane & 15;
  int hi = lane >> 4;
  int m0 = blockIdx.x << 4;
  int m = m0 + row;
  int b = m >> TLOG;
  int t = m & (TOUT - 1);
  const _Float16* Xb = Xh + (size_t)b * (TIN * CIN);

  const _Float16* bp =
      packB + (((size_t)blockIdx.y * nKc) * 32 + lane) * 16;

  v8f acc = {};
#pragma unroll
  for (int c = 0; c < nKc; ++c) {
    __builtin_prefetch((const void*)(bp + 512), 0, 1);
    v16h bfrag = *(const v16h*)bp;
    bp += 512;
    v16h afrag;
#pragma unroll
    for (int g = 0; g < 2; ++g) {
      int base = c * 32 + g * 16 + (hi << 3);  // 8-aligned K run
      int ci = base & (CIN - 1);
      int k = base >> CLOG;                    // constant over the run
      int tin, ok;
      if constexpr (!TRANS) {
        tin = t * STRIDE + k - PAD;
        ok = (tin >= 0) & (tin < TIN);
      } else {
        int tn = t + PAD - k;
        if constexpr (STRIDE == 1) {
          tin = tn;
          ok = (tn >= 0) & (tn < TIN);
        } else {
          tin = tn >> 1;
          ok = (tn >= 0) & ((tn & 1) == 0) & (tin < TIN);
        }
      }
      const v8h* p = ok ? (const v8h*)(Xb + (size_t)tin * CIN + ci)
                        : (const v8h*)zrow;    // pointer select, no branch
      v8h r = *p;
#pragma unroll
      for (int e = 0; e < 8; ++e) afrag[g * 8 + e] = r[e];
    }
    acc = __builtin_amdgcn_wmma_f32_16x16x32_f16(
        false, afrag, false, bfrag, (short)0, acc, false, false);
  }

  int n = (blockIdx.y << 4) + row;
  float bb = bias[n];
#pragma unroll
  for (int v = 0; v < 8; ++v) {
    int mm = m0 + hi * 8 + v;
    int ob = mm >> TLOG;
    int ot = mm & (TOUT - 1);
    float yv = acc[v] + bb;
    Y[((size_t)ob * COUT + n) * TOUT + ot] = yv;
    if constexpr (WH)
      Yh[((size_t)ob * TOUT + ot) * COUT + n] = (_Float16)yv;
  }
}

// ---------------------------------------------------------------------------
// BatchNorm1d: batch stats over (N, L); apply (+relu / +residual) and emit
// the f16 channels-last companion for the next conv.
// ---------------------------------------------------------------------------
__global__ __launch_bounds__(256) void bn_stats(
    const float* __restrict__ X, float* __restrict__ mean,
    float* __restrict__ var, int Bn, int C, int Tn, int tLog) {
  int c = blockIdx.x;
  long NT = (long)Bn * Tn;
  float s = 0.f, s2 = 0.f;
  for (long i = threadIdx.x; i < NT; i += blockDim.x) {
    long b = i >> tLog;
    long t = i & (Tn - 1);
    float x = X[(b * C + c) * Tn + t];
    s += x;
    s2 += x * x;
  }
  __shared__ float sh[256];
  __shared__ float sh2[256];
  sh[threadIdx.x] = s;
  sh2[threadIdx.x] = s2;
  __syncthreads();
  for (int o = 128; o > 0; o >>= 1) {
    if ((int)threadIdx.x < o) {
      sh[threadIdx.x] += sh[threadIdx.x + o];
      sh2[threadIdx.x] += sh2[threadIdx.x + o];
    }
    __syncthreads();
  }
  if (threadIdx.x == 0) {
    float m = sh[0] / (float)NT;
    mean[c] = m;
    var[c] = sh2[0] / (float)NT - m * m;
  }
}

__global__ __launch_bounds__(256) void bn_apply(
    const float* __restrict__ X, const float* __restrict__ mean,
    const float* __restrict__ var, const float* __restrict__ g,
    const float* __restrict__ beta, const float* __restrict__ res,
    float* __restrict__ Y, _Float16* __restrict__ Yh, int cLog, int tLog,
    long total, int relu) {
  long i = blockIdx.x * (long)blockDim.x + threadIdx.x;
  if (i >= total) return;
  int t = (int)i & ((1 << tLog) - 1);
  int c = (int)(i >> tLog) & ((1 << cLog) - 1);
  long b = i >> (tLog + cLog);
  float y = (X[i] - mean[c]) * rsqrtf(var[c] + 1e-5f) * g[c] + beta[c];
  if (relu) y = fmaxf(y, 0.f);
  if (res) y += res[i];
  Y[i] = y;
  Yh[((size_t)b << (tLog + cLog)) + ((size_t)t << cLog) + c] = (_Float16)y;
}

// ---------------------------------------------------------------------------
// LSTM: gates GEMM (A = [x_t | h_{t-1}], K = CIN+HH) then pointwise cell.
// Xh layout f16 [B, LT, CIN]; H16 layout f16 [B, HH].
// ---------------------------------------------------------------------------
template <int CIN, int HH, int LT>
__global__ __launch_bounds__(32) void lstm_gates_wmma(
    const _Float16* __restrict__ Xh, const _Float16* __restrict__ H16,
    const _Float16* __restrict__ packB, const float* __restrict__ bias,
    float* __restrict__ gates, int t) {
  constexpr int Ktot = CIN + HH;
  constexpr int nKc = Ktot / 32;
  int lane = threadIdx.x & 31;
  int row = lane & 15;
  int hi = lane >> 4;
  int m0 = blockIdx.x << 4;
  int m = m0 + row;

  const _Float16* xrow = Xh + ((size_t)m * LT + t) * CIN;
  const _Float16* hrow = H16 + (size_t)m * HH;
  const _Float16* bp =
      packB + (((size_t)blockIdx.y * nKc) * 32 + lane) * 16;

  v8f acc = {};
#pragma unroll
  for (int c = 0; c < nKc; ++c) {
    __builtin_prefetch((const void*)(bp + 512), 0, 1);
    v16h bfrag = *(const v16h*)bp;
    bp += 512;
    v16h afrag;
#pragma unroll
    for (int g = 0; g < 2; ++g) {
      int base = c * 32 + g * 16 + (hi << 3);
      const _Float16* src;
      if (c * 32 + g * 16 < CIN)   // whole run from x_t (folds at compile)
        src = xrow + base;
      else
        src = hrow + (base - CIN);
      v8h r = *(const v8h*)src;
#pragma unroll
      for (int e = 0; e < 8; ++e) afrag[g * 8 + e] = r[e];
    }
    acc = __builtin_amdgcn_wmma_f32_16x16x32_f16(
        false, afrag, false, bfrag, (short)0, acc, false, false);
  }

  int n = (blockIdx.y << 4) + row;
  float bb = bias[n];
#pragma unroll
  for (int v = 0; v < 8; ++v) {
    int mm = m0 + hi * 8 + v;
    gates[(size_t)mm * (4 * HH) + n] = acc[v] + bb;
  }
}

__global__ __launch_bounds__(256) void lstm_cell(
    const float* __restrict__ gates, _Float16* __restrict__ h16,
    float* __restrict__ cbuf, float* __restrict__ Yseq,
    _Float16* __restrict__ Yh16, int t) {
  constexpr int HH = 128, LT = 16;
  int i = blockIdx.x * blockDim.x + threadIdx.x;  // over B*H = 131072
  int b = i >> 7;
  int j = i & (HH - 1);
  const float* gr = gates + (size_t)b * 4 * HH;
  float ig = gr[j];
  float fg = gr[HH + j];
  float gg = gr[2 * HH + j];
  float og = gr[3 * HH + j];
  float c = cbuf[i];
  float si = 1.f / (1.f + expf(-ig));
  float sf = 1.f / (1.f + expf(-fg));
  float so = 1.f / (1.f + expf(-og));
  c = sf * c + si * tanhf(gg);
  float h = so * tanhf(c);
  cbuf[i] = c;
  h16[i] = (_Float16)h;
  Yseq[((size_t)b * HH + j) * LT + t] = h;                 // [B,C,L] f32
  Yh16[((size_t)b * LT + t) * HH + j] = (_Float16)h;       // [B,L,C] f16
}

// ---------------------------------------------------------------------------
// VQ: argmin_j (||w_j||^2 - 2 z.w_j) via WMMA fused with running min, then
// gather (f16 channels-last for lstm2) + loss accumulation.
// ---------------------------------------------------------------------------
__global__ __launch_bounds__(256) void vq_code_norms(
    const float* __restrict__ codes, float* __restrict__ ww) {
  int j = blockIdx.x * blockDim.x + threadIdx.x;
  if (j >= 512) return;
  float s = 0.f;
#pragma unroll
  for (int d = 0; d < 16; ++d) {
    float v = codes[j * 16 + d];
    s += v * v;
  }
  ww[j] = s;
}

__global__ __launch_bounds__(32) void vq_argmin_wmma(
    const float* __restrict__ Z, const _Float16* __restrict__ packC,
    const float* __restrict__ ww, int* __restrict__ idx) {
  int lane = threadIdx.x & 31;
  int row = lane & 15;
  int hi = lane >> 4;
  int m0 = blockIdx.x << 4;

  // z row is 16 contiguous floats; K halfwords 0..15 live, 16..31 zero.
  v16h afrag;
  const float* zr = Z + (size_t)(m0 + row) * 16;
#pragma unroll
  for (int h = 0; h < 16; ++h) {
    int g = (h >> 1) >> 2;
    int w2 = ((h >> 1) & 3) * 2 + (h & 1);
    afrag[h] = (g == 0) ? (_Float16)zr[hi * 8 + w2] : (_Float16)0.f;
  }

  float bestv[8];
  int besti[8];
#pragma unroll
  for (int v = 0; v < 8; ++v) {
    bestv[v] = 3.4e38f;
    besti[v] = 0;
  }

#pragma unroll 8
  for (int jt = 0; jt < 32; ++jt) {
    const _Float16* bp = packC + ((size_t)jt * 32 + lane) * 16;
    v16h bfrag = *(const v16h*)bp;
    v8f acc = {};
    acc = __builtin_amdgcn_wmma_f32_16x16x32_f16(
        false, afrag, false, bfrag, (short)0, acc, false, false);
    int j = (jt << 4) + row;
    float wwj = ww[j];
#pragma unroll
    for (int v = 0; v < 8; ++v) {
      float s = wwj - 2.f * acc[v];
      if (s < bestv[v]) {
        bestv[v] = s;
        besti[v] = j;
      }
    }
  }

#pragma unroll
  for (int v = 0; v < 8; ++v) {
    float mv = bestv[v];
    int mi = besti[v];
#pragma unroll
    for (int off = 1; off < 16; off <<= 1) {
      float ov = __shfl_xor(mv, off, 32);
      int oi = __shfl_xor(mi, off, 32);
      if (ov < mv || (ov == mv && oi < mi)) {
        mv = ov;
        mi = oi;
      }
    }
    if (row == 0) idx[m0 + hi * 8 + v] = mi;
  }
}

__global__ __launch_bounds__(256) void vq_gather_loss(
    const float* __restrict__ Z, const int* __restrict__ idx,
    const float* __restrict__ codes, _Float16* __restrict__ D1h,
    float* __restrict__ lossAcc, int Nrows) {
  int n = blockIdx.x * blockDim.x + threadIdx.x;  // n = b*128 + cch
  float s = 0.f;
  if (n < Nrows) {
    int j = idx[n];
    int b = n >> 7;
    int cch = n & 127;
    const float* cj = codes + j * 16;
    const float* zr = Z + (size_t)n * 16;
#pragma unroll
    for (int d = 0; d < 16; ++d) {
      float wv = cj[d];
      float df = zr[d] - wv;
      s += df * df;
      D1h[((size_t)b * 16 + d) * 128 + cch] = (_Float16)wv;  // [B,L,C] f16
    }
  }
  __shared__ float sh[256];
  sh[threadIdx.x] = s;
  __syncthreads();
  for (int o = 128; o > 0; o >>= 1) {
    if ((int)threadIdx.x < o) sh[threadIdx.x] += sh[threadIdx.x + o];
    __syncthreads();
  }
  if (threadIdx.x == 0) atomicAdd(lossAcc, sh[0]);
}

__global__ void finalize_losses(const float* __restrict__ lossAcc,
                                float* __restrict__ out2, float invN) {
  // forward values of commit and embed losses are identical
  out2[0] = lossAcc[0] * invN;
  out2[1] = lossAcc[0] * invN;
}

__global__ __launch_bounds__(256) void zero_f32(float* __restrict__ p,
                                                long n) {
  long i = blockIdx.x * (long)blockDim.x + threadIdx.x;
  if (i < n) p[i] = 0.f;
}

// ---------------------------------------------------------------------------
// Host orchestration
// ---------------------------------------------------------------------------
namespace {

inline size_t pack_halves(int Cout, int Cin, int KW) {
  int Kt = Cin * KW;
  return (size_t)((Cout + 15) / 16) * ((Kt + 31) / 32) * 512;
}

inline void launch_pack(const float* W, _Float16* pack, int Cout, int Cin,
                        int KW, int mode, hipStream_t s) {
  int total = (int)pack_halves(Cout, Cin, KW);
  pack_conv_weights<<<(total + 255) / 256, 256, 0, s>>>(W, pack, Cout, Cin,
                                                        KW, mode);
}

}  // namespace

extern "C" void kernel_launch(void* const* d_in, const int* in_sizes, int n_in,
                              void* d_out, int out_size, void* d_ws,
                              size_t ws_size, hipStream_t stream) {
  (void)in_sizes; (void)out_size; (void)ws_size;

  constexpr int B = 1024, ORG = 32, T = 64, C1 = 64, C2 = 128, H = 128;
  constexpr int L = 16, VQ_NUM = 512;
  constexpr int NROWS = B * C2;  // 131072 VQ rows

  const float* IN[66];
  for (int i = 0; i < 66 && i < n_in; ++i) IN[i] = (const float*)d_in[i];
  // d_in order (setup_inputs insertion order):
  // 0:x 1-4:conv0{w,b,g,beta} 5-12:res1[0..1] 13-14:ds1 15-16:ch12
  // 17-28:res2[0..2] 29-30:ds2 31-33:lstm1 34:embed 35-37:lstm2 38-39:us2
  // 40-51:deres2[0..2] 52-53:ch21 54-55:us1 56-63:deres1[0..1] 64-65:deconv

  size_t off = 0;
  auto allocf = [&](size_t n) -> float* {
    float* p = (float*)d_ws + off;
    off += (n + 31) & ~(size_t)31;
    return p;
  };
  auto alloch = [&](size_t nh) -> _Float16* {
    return (_Float16*)allocf((nh + 1) / 2);
  };

  float* bufA = allocf(4194304);
  float* bufB = allocf(4194304);
  float* bufC = allocf(4194304);
  float* bufD = allocf(4194304);
  float* sm1 = allocf(2097152);
  float* sm2 = allocf(2097152);
  float* gates = allocf((size_t)B * 4 * H);
  float* cbuf = allocf((size_t)B * H);
  float* bmean = allocf(128);
  float* bvar = allocf(128);
  float* ww = allocf(VQ_NUM);
  float* lossAcc = allocf(32);
  int* vqIdx = (int*)allocf(NROWS);

  // f16 activation companions (channels-last [B,T,C])
  _Float16* xh0 = alloch((size_t)B * T * ORG);
  _Float16* bnh = alloch((size_t)B * 64 * 64);  // 4M halves, shared by BNs
  _Float16* ds1h = alloch((size_t)B * 32 * C1);
  _Float16* ds2h = alloch((size_t)B * 16 * C2);
  _Float16* us2h = alloch((size_t)B * 32 * C2);
  _Float16* us1h = alloch((size_t)B * 64 * C1);
  _Float16* lstmYh = alloch((size_t)B * L * C2);
  _Float16* d1h = alloch((size_t)B * L * C2);
  _Float16* h16 = alloch((size_t)B * H);
  _Float16* zrow = alloch(128);  // zero guard row (>= max CIN)

  auto allocPack = [&](int Cout, int Cin, int KW) -> _Float16* {
    return (_Float16*)allocf(pack_halves(Cout, Cin, KW) / 2);
  };

  _Float16* pConv0 = allocPack(C1, ORG, 5);
  _Float16* pRes1a = allocPack(C1, C1, 3);
  _Float16* pRes1b = allocPack(C1, C1, 3);
  _Float16* pDs1 = allocPack(C1, C1, 2);
  _Float16* pCh12 = allocPack(C2, C1, 1);
  _Float16* pRes2a = allocPack(C2, C1, 3);
  _Float16* pRes2b = allocPack(C2, C2, 3);
  _Float16* pRes2c = allocPack(C2, C2, 3);
  _Float16* pDs2 = allocPack(C2, C2, 2);
  _Float16* pUs2 = allocPack(C2, C2, 2);
  _Float16* pDer2a = allocPack(C1, C2, 3);
  _Float16* pDer2b = allocPack(C1, C1, 3);
  _Float16* pDer2c = allocPack(C1, C1, 3);
  _Float16* pCh21 = allocPack(C1, C2, 1);
  _Float16* pUs1 = allocPack(C1, C1, 2);
  _Float16* pDer1a = allocPack(C1, C1, 5);
  _Float16* pDer1b = allocPack(C1, C1, 5);
  _Float16* pDeconv = allocPack(ORG, C1, 5);
  _Float16* packL1 = (_Float16*)allocf(pack_halves(4 * H, C2 + H, 1) / 2);
  _Float16* packL2 = (_Float16*)allocf(pack_halves(4 * H, C2 + H, 1) / 2);
  _Float16* packVQ = (_Float16*)allocf(pack_halves(VQ_NUM, 16, 1) / 2);

  // ---- prep: packs, norms, zeros, input conversion ----
  launch_pack(IN[1], pConv0, C1, ORG, 5, 0, stream);
  launch_pack(IN[5], pRes1a, C1, C1, 3, 0, stream);
  launch_pack(IN[9], pRes1b, C1, C1, 3, 0, stream);
  launch_pack(IN[13], pDs1, C1, C1, 2, 0, stream);
  launch_pack(IN[15], pCh12, C2, C1, 1, 0, stream);
  launch_pack(IN[17], pRes2a, C2, C1, 3, 0, stream);
  launch_pack(IN[21], pRes2b, C2, C2, 3, 0, stream);
  launch_pack(IN[25], pRes2c, C2, C2, 3, 0, stream);
  launch_pack(IN[29], pDs2, C2, C2, 2, 0, stream);
  launch_pack(IN[38], pUs2, C2, C2, 2, 1, stream);
  launch_pack(IN[40], pDer2a, C1, C2, 3, 1, stream);
  launch_pack(IN[44], pDer2b, C1, C1, 3, 1, stream);
  launch_pack(IN[48], pDer2c, C1, C1, 3, 1, stream);
  launch_pack(IN[52], pCh21, C1, C2, 1, 0, stream);
  launch_pack(IN[54], pUs1, C1, C1, 2, 1, stream);
  launch_pack(IN[56], pDer1a, C1, C1, 5, 1, stream);
  launch_pack(IN[60], pDer1b, C1, C1, 5, 1, stream);
  launch_pack(IN[64], pDeconv, ORG, C1, 5, 1, stream);
  {
    int tot = (int)pack_halves(4 * H, C2 + H, 1);
    pack_lstm_weights<<<(tot + 255) / 256, 256, 0, stream>>>(IN[31], IN[32],
                                                             packL1, C2, H);
    pack_lstm_weights<<<(tot + 255) / 256, 256, 0, stream>>>(IN[35], IN[36],
                                                             packL2, C2, H);
    launch_pack(IN[34], packVQ, VQ_NUM, 16, 1, 0, stream);
    vq_code_norms<<<2, 256, 0, stream>>>(IN[34], ww);
    zero_f32<<<1, 96, 0, stream>>>(lossAcc, 32);
    zero_f32<<<1, 64, 0, stream>>>((float*)zrow, 64);
    long tx = (long)B * ORG * T;
    convert_nct_ntc<<<(int)(tx / 256), 256, 0, stream>>>(IN[0], xh0,
                                                         clog2(ORG),
                                                         clog2(T), tx);
  }

#define RUN_CONV(CIN, KW, S, P, TR, TIN, TOUT, COUT, WH, Xp, Yp, Yhp, packp, \
                 biasp)                                                       \
  conv_gemm_wmma<CIN, KW, S, P, TR, TIN, TOUT, COUT, WH>                      \
      <<<dim3((B * (TOUT)) / 16, (COUT) / 16), 32, 0, stream>>>(              \
          Xp, zrow, packp, biasp, Yp, Yhp)

  auto run_bn = [&](const float* X, float* Y, const float* g,
                    const float* beta, const float* res, int C, int Tn,
                    int relu) {
    bn_stats<<<C, 256, 0, stream>>>(X, bmean, bvar, B, C, Tn, clog2(Tn));
    long total = (long)B * C * Tn;
    bn_apply<<<(int)((total + 255) / 256), 256, 0, stream>>>(
        X, bmean, bvar, g, beta, res, Y, bnh, clog2(C), clog2(Tn), total,
        relu);
  };

  auto run_lstm = [&](const _Float16* Xseq16, float* Yseq, _Float16* Yh16,
                      _Float16* pack, const float* bias) {
    zero_f32<<<(B * H / 2 + 255) / 256, 256, 0, stream>>>((float*)h16,
                                                          B * H / 2);
    zero_f32<<<(B * H + 255) / 256, 256, 0, stream>>>(cbuf, B * H);
    for (int t = 0; t < L; ++t) {
      lstm_gates_wmma<C2, H, L>
          <<<dim3(B / 16, (4 * H) / 16), 32, 0, stream>>>(Xseq16, h16, pack,
                                                          bias, gates, t);
      lstm_cell<<<(B * H) / 256, 256, 0, stream>>>(gates, h16, cbuf, Yseq,
                                                   Yh16, t);
    }
  };

  // ================= encoder =================
  RUN_CONV(32, 5, 1, 2, 0, 64, 64, 64, 0, xh0, bufA, (_Float16*)0, pConv0,
           IN[2]);
  run_bn(bufA, bufB, IN[3], IN[4], nullptr, C1, T, 0);                      // h
  RUN_CONV(64, 3, 1, 1, 0, 64, 64, 64, 0, bnh, bufA, (_Float16*)0, pRes1a,
           IN[6]);
  run_bn(bufA, bufA, IN[7], IN[8], nullptr, C1, T, 1);
  RUN_CONV(64, 3, 1, 1, 0, 64, 64, 64, 0, bnh, bufC, (_Float16*)0, pRes1b,
           IN[10]);
  run_bn(bufC, bufC, IN[11], IN[12], bufB, C1, T, 0);                       // r1
  RUN_CONV(64, 2, 2, 0, 0, 64, 32, 64, 1, bnh, sm1, ds1h, pDs1, IN[14]);    // ds1
  RUN_CONV(64, 1, 1, 0, 0, 32, 32, 128, 0, ds1h, bufA, (_Float16*)0, pCh12,
           IN[16]);                                                         // ch12
  RUN_CONV(64, 3, 1, 1, 0, 32, 32, 128, 0, ds1h, bufB, (_Float16*)0, pRes2a,
           IN[18]);
  run_bn(bufB, bufB, IN[19], IN[20], nullptr, C2, T / 2, 1);
  RUN_CONV(128, 3, 1, 1, 0, 32, 32, 128, 0, bnh, bufC, (_Float16*)0, pRes2b,
           IN[22]);
  run_bn(bufC, bufC, IN[23], IN[24], nullptr, C2, T / 2, 1);
  RUN_CONV(128, 3, 1, 1, 0, 32, 32, 128, 0, bnh, bufD, (_Float16*)0, pRes2c,
           IN[26]);
  run_bn(bufD, bufD, IN[27], IN[28], bufA, C2, T / 2, 0);                   // r2
  RUN_CONV(128, 2, 2, 0, 0, 32, 16, 128, 1, bnh, sm1, ds2h, pDs2, IN[30]);  // e0
  run_lstm(ds2h, sm2, lstmYh, packL1, IN[33]);                              // z_e

  // ================= VQ =================
  vq_argmin_wmma<<<NROWS / 16, 32, 0, stream>>>(sm2, packVQ, ww, vqIdx);
  vq_gather_loss<<<NROWS / 256, 256, 0, stream>>>(sm2, vqIdx, IN[34], d1h,
                                                  lossAcc, NROWS);

  // ================= decoder =================
  run_lstm(d1h, sm1, lstmYh, packL2, IN[37]);                               // D2
  RUN_CONV(128, 2, 2, 0, 1, 16, 32, 128, 1, lstmYh, bufA, us2h, pUs2,
           IN[39]);                                                         // u2
  RUN_CONV(128, 1, 1, 0, 0, 32, 32, 64, 0, us2h, sm2, (_Float16*)0, pCh21,
           IN[53]);                                                         // ch21
  RUN_CONV(128, 3, 1, 1, 1, 32, 32, 64, 0, us2h, bufB, (_Float16*)0, pDer2a,
           IN[41]);
  run_bn(bufB, bufB, IN[42], IN[43], nullptr, C1, T / 2, 1);
  RUN_CONV(64, 3, 1, 1, 1, 32, 32, 64, 0, bnh, bufC, (_Float16*)0, pDer2b,
           IN[45]);
  run_bn(bufC, bufC, IN[46], IN[47], nullptr, C1, T / 2, 1);
  RUN_CONV(64, 3, 1, 1, 1, 32, 32, 64, 0, bnh, bufB, (_Float16*)0, pDer2c,
           IN[49]);
  run_bn(bufB, bufB, IN[50], IN[51], sm2, C1, T / 2, 0);                    // dr2
  RUN_CONV(64, 2, 2, 0, 1, 32, 64, 64, 1, bnh, bufA, us1h, pUs1, IN[55]);   // u1
  RUN_CONV(64, 5, 1, 2, 1, 64, 64, 64, 0, us1h, bufC, (_Float16*)0, pDer1a,
           IN[57]);
  run_bn(bufC, bufC, IN[58], IN[59], nullptr, C1, T, 1);
  RUN_CONV(64, 5, 1, 2, 1, 64, 64, 64, 0, bnh, bufD, (_Float16*)0, pDer1b,
           IN[61]);
  run_bn(bufD, bufD, IN[62], IN[63], bufA, C1, T, 0);                       // dr1
  RUN_CONV(64, 5, 1, 2, 1, 64, 64, 32, 0, bnh, (float*)d_out, (_Float16*)0,
           pDeconv, IN[65]);                                                // recon

  finalize_losses<<<1, 1, 0, stream>>>(lossAcc, (float*)d_out + B * ORG * T,
                                       1.f / (float)NROWS);
#undef RUN_CONV
}
